// MultiHeadAttention_13829794693285
// MI455X (gfx1250) — matrix operand order repaired and checked by reading, not compiled, not run on hardware
//
#include <hip/hip_runtime.h>
#include <math.h>

// ---------------------------------------------------------------------------
// MI455X (gfx1250) Multi-Head Attention, bf16 WMMA (v_wmma_f32_16x16x32_bf16)
// Register-blocked GEMM: 8 waves/block as 2(M) x 4(N); each wave owns a
// 32 x (16*NWT) output tile = 2*NWT accumulators -> 8 WMMAs per 12 b128 loads.
// ---------------------------------------------------------------------------
// d_out layout: [ out : 8*1024*2048 f32 ][ weights : 128*1024*1024 f32 ]

typedef __attribute__((ext_vector_type(16))) __bf16 v16bf;
typedef __attribute__((ext_vector_type(8)))  __bf16 v8bf;
typedef __attribute__((ext_vector_type(8)))  float  v8f;

// ---- WMMA fragment loaders (wave32 layouts per CDNA5 ISA 7.12.2) ----------
// A-matrix 16x32 bf16: lanes 0-15 -> row M=lane, halves = K {0..7,16..23};
//                      lanes 16-31 -> row M=lane-16, halves = K {8..15,24..31}
__device__ __forceinline__ v16bf frag_a_bf16(const __bf16* A, int lda,
                                             int l16, int hi, int k0) {
    const __bf16* p = A + (size_t)l16 * lda + k0 + hi * 8;
    v8bf lo = *(const v8bf*)(p);
    v8bf hh = *(const v8bf*)(p + 16);
    v16bf r;
#pragma unroll
    for (int i = 0; i < 8; ++i) { r[i] = lo[i]; r[i + 8] = hh[i]; }
    return r;
}

__device__ __forceinline__ v16bf frag_a_f32(const float* A, int lda,
                                            int l16, int hi, int k0) {
    const float* p = A + (size_t)l16 * lda + k0 + hi * 8;
    float t[16];
    *(float4*)(t + 0)  = *(const float4*)(p + 0);
    *(float4*)(t + 4)  = *(const float4*)(p + 4);
    *(float4*)(t + 8)  = *(const float4*)(p + 16);
    *(float4*)(t + 12) = *(const float4*)(p + 20);
    v16bf r;
#pragma unroll
    for (int i = 0; i < 16; ++i) r[i] = (__bf16)t[i];
    return r;
}

// B-matrix 32x16 bf16 for NT GEMM (B element (k,n) = W[n][k], W row-major):
// lane n = n0 + (lane&15); lanes 0-15 hold K {0..7,16..23}, lanes 16-31 K {8..15,24..31}
__device__ __forceinline__ v16bf frag_b_bf16(const __bf16* B, int ldb,
                                             int l16, int hi, int k0) {
    const __bf16* p = B + (size_t)l16 * ldb + k0 + hi * 8;
    v8bf lo = *(const v8bf*)(p);
    v8bf hh = *(const v8bf*)(p + 16);
    v16bf r;
#pragma unroll
    for (int i = 0; i < 8; ++i) { r[i] = lo[i]; r[i + 8] = hh[i]; }
    return r;
}

// ---- Register-blocked NT GEMM ---------------------------------------------
// C[m,n] = scale * sum_k A[m,k]*B[n,k] (+ bias[n]).  B is always bf16.
// blockDim=256 (8 waves = 2 wm x 4 wn). Wave tile: M=32 (2 tiles), N=16*NWT.
// Block tile: M=64, N=64*NWT.  grid = (N/(64*NWT), M/64, batch).
template <int NWT, bool A_F32, bool OUT_F32, bool BIAS>
__global__ __launch_bounds__(256) void
mha_gemm_nt(const void* __restrict__ Ap, const __bf16* __restrict__ Bp,
            const float* __restrict__ bias, void* __restrict__ Cp,
            int K, int lda, int ldb, int ldc,
            long long sA, long long sB, long long sC, float scale) {
    const int lane = threadIdx.x & 31;
    const int wave = threadIdx.x >> 5;
    const int l16  = lane & 15;
    const int hi   = lane >> 4;
    const int wm   = wave >> 2;           // 0..1
    const int wn   = wave & 3;            // 0..3
    const int m0   = blockIdx.y * 64 + wm * 32;
    const int n0   = blockIdx.x * (64 * NWT) + wn * (16 * NWT);
    const long long z = blockIdx.z;

    const float*  Af0 = (const float*)Ap  + z * sA + (size_t)m0 * lda;
    const __bf16* Ah0 = (const __bf16*)Ap + z * sA + (size_t)m0 * lda;
    const float*  Af1 = Af0 + (size_t)16 * lda;
    const __bf16* Ah1 = Ah0 + (size_t)16 * lda;
    const __bf16* Bh  = Bp + z * sB + (size_t)n0 * ldb;

    v8f acc[2][NWT] = {};

    for (int k0 = 0; k0 < K; k0 += 32) {
        v16bf a0 = A_F32 ? frag_a_f32(Af0, lda, l16, hi, k0)
                         : frag_a_bf16(Ah0, lda, l16, hi, k0);
        v16bf a1 = A_F32 ? frag_a_f32(Af1, lda, l16, hi, k0)
                         : frag_a_bf16(Ah1, lda, l16, hi, k0);
#pragma unroll
        for (int t = 0; t < NWT; ++t) {
            v16bf b = frag_b_bf16(Bh + (size_t)t * 16 * ldb, ldb, l16, hi, k0);
            acc[0][t] = __builtin_amdgcn_wmma_f32_16x16x32_bf16(
                false, a0, false, b, (short)0, acc[0][t], false, false);
            acc[1][t] = __builtin_amdgcn_wmma_f32_16x16x32_bf16(
                false, a1, false, b, (short)0, acc[1][t], false, false);
        }
    }

    float*  Cf = (float*)Cp  + z * sC;
    __bf16* Ch = (__bf16*)Cp + z * sC;
#pragma unroll
    for (int mi = 0; mi < 2; ++mi) {
#pragma unroll
        for (int t = 0; t < NWT; ++t) {
            const int col = n0 + t * 16 + l16;
            const float bv = BIAS ? bias[col] : 0.0f;
#pragma unroll
            for (int r = 0; r < 8; ++r) {
                const int row = m0 + mi * 16 + r + hi * 8;
                float v = acc[mi][t][r] * scale + bv;
                if (OUT_F32) Cf[(size_t)row * ldc + col] = v;
                else         Ch[(size_t)row * ldc + col] = (__bf16)v;
            }
        }
    }
}

// ---- f32 -> bf16 bulk convert (8 elems/thread, b128 in / b128 out) --------
__global__ __launch_bounds__(256) void mha_cvt_bf16(const float* __restrict__ s,
                                                    void* __restrict__ d) {
    const long long i = ((long long)blockIdx.x * 256 + threadIdx.x) * 8;
    float4 x = *(const float4*)(s + i);
    float4 y = *(const float4*)(s + i + 4);
    v8bf o;
    o[0] = (__bf16)x.x; o[1] = (__bf16)x.y; o[2] = (__bf16)x.z; o[3] = (__bf16)x.w;
    o[4] = (__bf16)y.x; o[5] = (__bf16)y.y; o[6] = (__bf16)y.z; o[7] = (__bf16)y.w;
    *(v8bf*)((__bf16*)d + i) = o;
}

// ---- Row softmax: one wave32 per row of 1024 (8 rows / block) -------------
__global__ __launch_bounds__(256) void mha_softmax_rows(float* __restrict__ w) {
    const int lane = threadIdx.x & 31;
    const int wave = threadIdx.x >> 5;
    float* p = w + ((long long)blockIdx.x * 8 + wave) * 1024;

    float v[32];
    float m = -INFINITY;
#pragma unroll
    for (int j = 0; j < 32; ++j) {
        v[j] = p[j * 32 + lane];
        m = fmaxf(m, v[j]);
    }
#pragma unroll
    for (int off = 16; off; off >>= 1) m = fmaxf(m, __shfl_xor(m, off, 32));
    float s = 0.0f;
#pragma unroll
    for (int j = 0; j < 32; ++j) { v[j] = __expf(v[j] - m); s += v[j]; }
#pragma unroll
    for (int off = 16; off; off >>= 1) s += __shfl_xor(s, off, 32);
    const float inv = 1.0f / s;
#pragma unroll
    for (int j = 0; j < 32; ++j) p[j * 32 + lane] = v[j] * inv;
}

// ---- Per-head transpose: VT[h][d][k] = Vb[h][k][d] (LDS tiled) ------------
__global__ __launch_bounds__(256) void mha_transpose_v(const void* __restrict__ Vp,
                                                       void* __restrict__ VTp) {
    __shared__ __bf16 tile[32][33];
    const __bf16* src = (const __bf16*)Vp + (size_t)blockIdx.z * 131072;
    __bf16*       dst = (__bf16*)VTp      + (size_t)blockIdx.z * 131072;
    const int k0 = blockIdx.x * 32;
    const int d0 = blockIdx.y * 32;
    const int tx = threadIdx.x & 31;
    const int ty = threadIdx.x >> 5;   // 0..7
#pragma unroll
    for (int i = 0; i < 32; i += 8)
        tile[ty + i][tx] = src[(size_t)(k0 + ty + i) * 128 + d0 + tx];
    __syncthreads();
#pragma unroll
    for (int i = 0; i < 32; i += 8)
        dst[(size_t)(d0 + ty + i) * 1024 + k0 + tx] = tile[tx][ty + i];
}

// ---------------------------------------------------------------------------
extern "C" void kernel_launch(void* const* d_in, const int* in_sizes, int n_in,
                              void* d_out, int out_size, void* d_ws, size_t ws_size,
                              hipStream_t stream) {
    (void)in_sizes; (void)n_in; (void)out_size; (void)ws_size;

    const float* query = (const float*)d_in[0];
    const float* keyi  = (const float*)d_in[1];
    const float* value = (const float*)d_in[2];
    const float* Wq = (const float*)d_in[3];  const float* Wqb = (const float*)d_in[4];
    const float* Wk = (const float*)d_in[5];  const float* Wkb = (const float*)d_in[6];
    const float* Wv = (const float*)d_in[7];  const float* Wvb = (const float*)d_in[8];
    const float* Wo = (const float*)d_in[9];  const float* Wob = (const float*)d_in[10];

    float* out = (float*)d_out;                 // 8*1024*2048
    float* wts = out + 16777216;                // 128*1024*1024

    char* ws = (char*)d_ws;
    void* Qb  = (void*)(ws + 0);                 // bf16 8192x2048 (33.5 MB)
    void* Kb  = (void*)(ws + 33554432ULL);
    void* Vb  = (void*)(ws + 67108864ULL);
    void* VT  = (void*)(ws + 100663296ULL);      // bf16 128 x [128,1024]
    void* Ctx = (void*)(ws + 134217728ULL);      // bf16 8192x2048
    void* WqB = (void*)(ws + 167772160ULL);      // bf16 2048x2048 (8.4 MB)
    void* WkB = (void*)(ws + 176160768ULL);
    void* WvB = (void*)(ws + 184549376ULL);
    void* WoB = (void*)(ws + 192937984ULL);      // total 201.3 MB

    const dim3 blk(256);
    const float inv_sqrt_d = 1.0f / sqrtf(2048.0f);

    // 0) One-time bf16 conversion of the four weight matrices (2048x2048)
    const dim3 gCvtW(4194304 / (256 * 8));
    mha_cvt_bf16<<<gCvtW, blk, 0, stream>>>(Wq, WqB);
    mha_cvt_bf16<<<gCvtW, blk, 0, stream>>>(Wk, WkB);
    mha_cvt_bf16<<<gCvtW, blk, 0, stream>>>(Wv, WvB);
    mha_cvt_bf16<<<gCvtW, blk, 0, stream>>>(Wo, WoB);

    // 1) Projections: [8192,2048] = X(f32, inline cvt) @ Wb(bf16)^T + b -> bf16
    const dim3 gProj(2048 / 256, 8192 / 64, 1);
    mha_gemm_nt<4, true, false, true><<<gProj, blk, 0, stream>>>(
        query, (const __bf16*)WqB, Wqb, Qb, 2048, 2048, 2048, 2048,
        0LL, 0LL, 0LL, 1.0f);
    mha_gemm_nt<4, true, false, true><<<gProj, blk, 0, stream>>>(
        keyi, (const __bf16*)WkB, Wkb, Kb, 2048, 2048, 2048, 2048,
        0LL, 0LL, 0LL, 1.0f);
    mha_gemm_nt<4, true, false, true><<<gProj, blk, 0, stream>>>(
        value, (const __bf16*)WvB, Wvb, Vb, 2048, 2048, 2048, 2048,
        0LL, 0LL, 0LL, 1.0f);

    // 2) Scores: per head [1024,1024] = Qh @ Kh^T * 1/sqrt(D) -> f32 (d_out)
    const dim3 gScore(1024 / 256, 1024 / 64, 128);
    mha_gemm_nt<4, false, true, false><<<gScore, blk, 0, stream>>>(
        Qb, (const __bf16*)Kb, nullptr, wts, 128, 128, 128, 1024,
        131072LL, 131072LL, 1048576LL, inv_sqrt_d);

    // 3) Softmax over last dim, in place (this IS the second output)
    mha_softmax_rows<<<dim3(131072 / 8), blk, 0, stream>>>(wts);

    // 4) Transpose V per head: [1024,128] -> [128,1024]
    mha_transpose_v<<<dim3(32, 4, 128), blk, 0, stream>>>(Vb, VT);

    // 5) Context: per head [1024,128] = P(f32, inline cvt) @ VT^T -> bf16
    const dim3 gCtx(1, 1024 / 64, 128);
    mha_gemm_nt<2, true, false, false><<<gCtx, blk, 0, stream>>>(
        wts, (const __bf16*)VT, nullptr, Ctx, 1024, 1024, 1024, 128,
        1048576LL, 131072LL, 131072LL, 1.0f);

    // 6) Output projection: [8192,2048] = Ctx(bf16) @ WoB(bf16)^T + b -> f32
    mha_gemm_nt<4, false, true, true><<<gProj, blk, 0, stream>>>(
        Ctx, (const __bf16*)WoB, Wob, out, 2048, 2048, 2048, 2048,
        0LL, 0LL, 0LL, 1.0f);
}
